// TokenMixer_85504208929081
// MI455X (gfx1250) — matrix operand order repaired and checked by reading, not compiled, not run on hardware
//
#include <hip/hip_runtime.h>
#include <math.h>

// Problem constants (match reference)
#define DIMN   1024
#define HEADS  16
#define HD     64
#define BATCH  2
#define SEQ    2048
#define ROWS   (BATCH * SEQ)   // 4096
#define EPSN   1e-5f
#define KC     64              // GEMM K-chunk staged in LDS

typedef __attribute__((ext_vector_type(16))) _Float16 v16h;
typedef __attribute__((ext_vector_type(8)))  _Float16 v8h;
typedef __attribute__((ext_vector_type(8)))  float    v8f;
typedef int v4i __attribute__((vector_size(16)));

union HU { v16h h16; v8h h8[2]; };

// -------- async global->LDS copy path (gfx1250), with safe fallback --------
#if defined(__HIP_DEVICE_COMPILE__) && \
    __has_builtin(__builtin_amdgcn_global_load_async_to_lds_b128) && \
    __has_builtin(__builtin_amdgcn_s_wait_asynccnt)
#define ASYNC_LDS 1
#else
#define ASYNC_LDS 0
#endif

__device__ __forceinline__ void cp_async16(_Float16* lds, const _Float16* g) {
#if ASYNC_LDS
  __builtin_amdgcn_global_load_async_to_lds_b128(
      (__attribute__((address_space(1))) v4i*)g,
      (__attribute__((address_space(3))) v4i*)lds,
      0, 0);
#else
  *(v8h*)lds = *(const v8h*)g;
#endif
}

#if ASYNC_LDS
#define WAIT_ASYNC(n) __builtin_amdgcn_s_wait_asynccnt(n)
#else
#define WAIT_ASYNC(n) do {} while (0)
#endif

// Load a 16x32 f16 A-fragment per ISA layout:
// lane(0-15)=row M, holds K {0..7,16..23}; lane(16-31)=row M, K {8..15,24..31}.
// `base` must already point at rowStart + k0 + hi*8 (16B aligned).
static __device__ __forceinline__ v16h load_afrag(const _Float16* base) {
  HU u;
  u.h8[0] = *(const v8h*)(base);
  u.h8[1] = *(const v8h*)(base + 16);
  return u.h16;
}

// ---------------------------------------------------------------------------
// Kernel: fp32 -> fp16 convert (grid-stride)
// ---------------------------------------------------------------------------
__global__ void k_f32_to_f16(const float* __restrict__ in,
                             _Float16* __restrict__ out, int n) {
  for (int i = blockIdx.x * blockDim.x + threadIdx.x; i < n;
       i += gridDim.x * blockDim.x)
    out[i] = (_Float16)in[i];
}

// ---------------------------------------------------------------------------
// Kernel: Y = A @ W^T   (A: [rows x 1024] f16, W: [1024 x 1024] f16 row-major,
// Y: [rows x 1024] f32).  Block = 256 thr = 8 waves; block tile 128x64.
// A/B chunks double-buffered in LDS via async copies (6 b128 per thread per
// chunk), wave tile 16x64 (4 accumulators).  grid = (DIMN/64, rows/128)
// ---------------------------------------------------------------------------
__global__ __launch_bounds__(256) void k_gemm(const _Float16* __restrict__ A,
                                              const _Float16* __restrict__ W,
                                              float* __restrict__ Y) {
  __shared__ _Float16 sA[2][128][KC];   // 2 x 16 KB
  __shared__ _Float16 sB[2][64][KC];    // 2 x  8 KB

  const int tid  = threadIdx.x;
  const int lane = tid & 31;
  const int wave = tid >> 5;
  const int lr   = lane & 15;
  const int hi   = (lane >> 4) & 1;
  const int col0 = blockIdx.x * 64;
  const int row0 = blockIdx.y * 128;

  const _Float16* Ab = A + (size_t)row0 * DIMN;
  const _Float16* Wb = W + (size_t)col0 * DIMN;

  // Issue one K-chunk's cooperative copy: A 128x64 (1024 x b128),
  // B 64x64 (512 x b128).  6 async ops per thread.
  auto issue_chunk = [&](int buf, int kc) {
#pragma unroll
    for (int i = 0; i < 4; ++i) {                 // A: id in [0,1024)
      const int id  = tid + i * 256;
      const int row = id >> 3, seg = id & 7;      // seg = 16B unit within row
      cp_async16(&sA[buf][row][seg * 8],
                 Ab + (size_t)row * DIMN + kc + seg * 8);
    }
#pragma unroll
    for (int i = 0; i < 2; ++i) {                 // B: id in [0,512)
      const int id  = tid + i * 256;
      const int row = id >> 3, seg = id & 7;
      cp_async16(&sB[buf][row][seg * 8],
                 Wb + (size_t)row * DIMN + kc + seg * 8);
    }
  };

  v8f acc[4] = {};
  issue_chunk(0, 0);

  for (int kc = 0, it = 0; kc < DIMN; kc += KC, ++it) {
    const int buf = it & 1;
    if (kc + KC < DIMN) {
      issue_chunk(buf ^ 1, kc + KC);
      WAIT_ASYNC(6);                  // previous chunk's 6 ops complete
    } else {
      WAIT_ASYNC(0);
    }
    __syncthreads();

#pragma unroll
    for (int k0 = 0; k0 < KC; k0 += 32) {
      v16h Af = load_afrag(&sA[buf][wave * 16 + lr][k0 + hi * 8]);
#pragma unroll
      for (int c = 0; c < 4; ++c) {
        v16h Bf = load_afrag(&sB[buf][c * 16 + lr][k0 + hi * 8]);
        acc[c] = __builtin_amdgcn_wmma_f32_16x16x32_f16(
            false, Af, false, Bf, (short)0, acc[c], false, false);
      }
    }
    __syncthreads();  // buf free for reuse before next issue
  }

#pragma unroll
  for (int c = 0; c < 4; ++c)
#pragma unroll
    for (int r = 0; r < 8; ++r)
      Y[(size_t)(row0 + wave * 16 + r + hi * 8) * DIMN + col0 + c * 16 + lr] =
          acc[c][r];
}

// ---------------------------------------------------------------------------
// Kernel: hnorm epilogue + repack.  One block (256 thr) per row of Y.
//  t = y + bias; t *= 1/(||t||+eps)
//  mode 0: write f16 head-major  q/k: out[((b*16+h)*SEQ + n)*64 + d]
//  mode 1: write f16 transposed  v:   out[((b*16+h)*64 + d)*SEQ + n]
//  mode 2: write f32 row-major to outf (final output)
// ---------------------------------------------------------------------------
__global__ __launch_bounds__(256) void k_hnorm_pack(const float* __restrict__ Y,
                                                    const float* __restrict__ bias,
                                                    _Float16* __restrict__ outh,
                                                    float* __restrict__ outf,
                                                    int mode) {
  const int row = blockIdx.x;  // [0, ROWS)
  const float* y = Y + (size_t)row * DIMN;
  __shared__ float red[8];

  float s = 0.f;
  for (int j = threadIdx.x; j < DIMN; j += 256) {
    float t = y[j] + bias[j];
    s += t * t;
  }
#pragma unroll
  for (int off = 16; off >= 1; off >>= 1) s += __shfl_xor(s, off, 32);
  if ((threadIdx.x & 31) == 0) red[threadIdx.x >> 5] = s;
  __syncthreads();
  float tot = red[0] + red[1] + red[2] + red[3] +
              red[4] + red[5] + red[6] + red[7];
  const float inv = 1.f / (sqrtf(tot) + EPSN);

  const int b  = row / SEQ;
  const int nn = row % SEQ;
  for (int j = threadIdx.x; j < DIMN; j += 256) {
    float t = (y[j] + bias[j]) * inv;
    if (mode == 2) {
      outf[(size_t)row * DIMN + j] = t;
    } else {
      const int h = j >> 6, d = j & 63;
      const size_t bh = (size_t)(b * HEADS + h);
      if (mode == 0)
        outh[(bh * SEQ + nn) * HD + d] = (_Float16)t;
      else
        outh[(bh * HD + d) * SEQ + nn] = (_Float16)t;
    }
  }
}

// ---------------------------------------------------------------------------
// Kernel: causal flash-attention per (b,h).  Block = 128 thr = 4 waves.
// grid = (SEQ/64, BATCH*HEADS).  Wave w owns 16 query rows.  K/V tiles are
// staged in LDS via double-buffered async copies (4 b128 per thread per tile)
// and shared by all 4 waves.
//  qh,kh: [bh][n][64] f16;  vt: [bh][64][n] f16;  ah: [b][n][1024] f16.
// ---------------------------------------------------------------------------
__global__ __launch_bounds__(128) void k_attn(const _Float16* __restrict__ qh,
                                              const _Float16* __restrict__ kh,
                                              const _Float16* __restrict__ vt,
                                              _Float16* __restrict__ ah) {
  __shared__ _Float16 sK[2][32][64];   // 2 x 4 KB, [key][d]
  __shared__ _Float16 sV[2][64][32];   // 2 x 4 KB, [d][key]
  __shared__ _Float16 sP[4][16][32];   // per-wave P staging (D->A relayout)

  const int tid  = threadIdx.x;
  const int lane = tid & 31;
  const int wave = tid >> 5;
  const int lr   = lane & 15;
  const int hi   = (lane >> 4) & 1;
  const int qt   = blockIdx.x;        // query tile of 64
  const int bh   = blockIdx.y;        // b*16 + h
  const int b    = bh >> 4;
  const int h    = bh & 15;
  const int q0   = qt * 64 + wave * 16;

  const _Float16* qp = qh + (size_t)bh * SEQ * HD;
  const _Float16* kp = kh + (size_t)bh * SEQ * HD;
  const _Float16* vp = vt + (size_t)bh * HD * SEQ;

  // Cooperative async copy of one 32-key tile of K and V (4 ops/thread).
  auto issue_kv = [&](int buf, int kbase) {
#pragma unroll
    for (int i = 0; i < 2; ++i) {                 // K: 4KB contiguous
      const int id = tid + i * 128;               // [0,256)
      cp_async16(&sK[buf][0][0] + id * 8,
                 kp + (size_t)kbase * HD + id * 8);
    }
#pragma unroll
    for (int i = 0; i < 2; ++i) {                 // V: 64 rows x 64B
      const int id  = tid + i * 128;              // [0,256)
      const int row = id >> 2, seg = id & 3;
      cp_async16(&sV[buf][row][seg * 8],
                 vp + (size_t)row * SEQ + kbase + seg * 8);
    }
  };

  // Q fragments (contraction d split 0..31 / 32..63), kept in registers.
  v16h Qf[2];
#pragma unroll
  for (int dc = 0; dc < 2; ++dc)
    Qf[dc] = load_afrag(qp + (size_t)(q0 + lr) * HD + dc * 32 + hi * 8);

  v8f O[4] = {};
  float m[8], l[8];
#pragma unroll
  for (int r = 0; r < 8; ++r) { m[r] = -3.0e38f; l[r] = 0.f; }

  const int ktEnd = 2 * qt + 2;  // key tiles of 32 touching this query tile
  issue_kv(0, 0);

  for (int kt = 0; kt < ktEnd; ++kt) {
    const int kbase = kt * 32;
    const int buf   = kt & 1;
    if (kt + 1 < ktEnd) {
      issue_kv(buf ^ 1, (kt + 1) * 32);
      WAIT_ASYNC(4);               // current tile's 4 ops complete
    } else {
      WAIT_ASYNC(0);
    }
    __syncthreads();

    // S = Q @ K^T : two 16-key halves, contraction 64 = 2 k-steps each
    v8f S[2] = {};
#pragma unroll
    for (int sh = 0; sh < 2; ++sh) {
#pragma unroll
      for (int dc = 0; dc < 2; ++dc) {
        v16h Bf = load_afrag(&sK[buf][sh * 16 + lr][dc * 32 + hi * 8]);
        S[sh] = __builtin_amdgcn_wmma_f32_16x16x32_f16(
            false, Qf[dc], false, Bf, (short)0, S[sh], false, false);
      }
    }

    // Online softmax.  D-layout: element r of lane = (row q0+r+hi*8, key lr).
#pragma unroll
    for (int r = 0; r < 8; ++r) {
      const int qrow = q0 + r + hi * 8;
      float s0 = S[0][r] * 0.125f;                 // 1/sqrt(64)
      float s1 = S[1][r] * 0.125f;
      if (kbase + lr > qrow)      s0 = -3.0e38f;   // causal mask
      if (kbase + 16 + lr > qrow) s1 = -3.0e38f;

      float mt = fmaxf(s0, s1);
#pragma unroll
      for (int off = 1; off < 16; off <<= 1)       // 16-lane row reduction
        mt = fmaxf(mt, __shfl_xor(mt, off, 32));
      const float mnew  = fmaxf(m[r], mt);
      const float alpha = __expf(m[r] - mnew);
      const float p0 = __expf(s0 - mnew);
      const float p1 = __expf(s1 - mnew);
      float ps = p0 + p1;
#pragma unroll
      for (int off = 1; off < 16; off <<= 1)
        ps += __shfl_xor(ps, off, 32);
      l[r] = l[r] * alpha + ps;
      m[r] = mnew;
#pragma unroll
      for (int c = 0; c < 4; ++c) O[c][r] *= alpha;

      sP[wave][r + hi * 8][lr]      = (_Float16)p0;
      sP[wave][r + hi * 8][16 + lr] = (_Float16)p1;
    }

    // Reload P as a 16x32 A-fragment (wave-local LDS; DS ops are in-order).
    v16h Pf = load_afrag(&sP[wave][lr][hi * 8]);

    // O += P @ V  (V transposed in LDS: contraction keys contiguous)
#pragma unroll
    for (int c = 0; c < 4; ++c) {
      v16h Vf = load_afrag(&sV[buf][c * 16 + lr][hi * 8]);
      O[c] = __builtin_amdgcn_wmma_f32_16x16x32_f16(
          false, Pf, false, Vf, (short)0, O[c], false, false);
    }
    __syncthreads();  // buf free before next issue
  }

  // Epilogue: divide by row sums, scatter into [b][n][DIM] f16.
#pragma unroll
  for (int c = 0; c < 4; ++c)
#pragma unroll
    for (int r = 0; r < 8; ++r) {
      const int n = q0 + r + hi * 8;
      const float o = O[c][r] / l[r];
      ah[((size_t)(b * SEQ + n)) * DIMN + h * HD + c * 16 + lr] = (_Float16)o;
    }
}

// ---------------------------------------------------------------------------
// Host launcher.  Workspace usage: 64 MB.
// ---------------------------------------------------------------------------
extern "C" void kernel_launch(void* const* d_in, const int* in_sizes, int n_in,
                              void* d_out, int out_size, void* d_ws, size_t ws_size,
                              hipStream_t stream) {
  (void)in_sizes; (void)n_in; (void)out_size; (void)ws_size;

  const float* x  = (const float*)d_in[0];
  const float* Wq = (const float*)d_in[1];
  const float* bq = (const float*)d_in[2];
  const float* Wk = (const float*)d_in[3];
  const float* bk = (const float*)d_in[4];
  const float* Wv = (const float*)d_in[5];
  const float* bv = (const float*)d_in[6];
  const float* Wo = (const float*)d_in[7];
  const float* bo = (const float*)d_in[8];
  // d_in[9] (mask) ignored: causal mask applied analytically.

  const size_t NE = (size_t)ROWS * DIMN;  // 4M elements
  char* p = (char*)d_ws;
  _Float16* xh  = (_Float16*)p; p += NE * 2;                  //  8 MB
  _Float16* Wqh = (_Float16*)p; p += (size_t)DIMN * DIMN * 2;
  _Float16* Wkh = (_Float16*)p; p += (size_t)DIMN * DIMN * 2;
  _Float16* Wvh = (_Float16*)p; p += (size_t)DIMN * DIMN * 2;
  _Float16* Woh = (_Float16*)p; p += (size_t)DIMN * DIMN * 2; // +8 MB
  float*    yraw = (float*)p;   p += NE * 4;                  // +16 MB (reused)
  _Float16* qh  = (_Float16*)p; p += NE * 2;                  // +8 MB
  _Float16* khb = (_Float16*)p; p += NE * 2;                  // +8 MB
  _Float16* vtb = (_Float16*)p; p += NE * 2;                  // +8 MB
  _Float16* ahb = (_Float16*)p; p += NE * 2;                  // +8 MB => 64 MB

  // 1) fp32 -> fp16 converts
  k_f32_to_f16<<<2048, 256, 0, stream>>>(x,  xh,  (int)NE);
  k_f32_to_f16<<<1024, 256, 0, stream>>>(Wq, Wqh, DIMN * DIMN);
  k_f32_to_f16<<<1024, 256, 0, stream>>>(Wk, Wkh, DIMN * DIMN);
  k_f32_to_f16<<<1024, 256, 0, stream>>>(Wv, Wvh, DIMN * DIMN);
  k_f32_to_f16<<<1024, 256, 0, stream>>>(Wo, Woh, DIMN * DIMN);

  const dim3 ggrid(DIMN / 64, ROWS / 128);  // (16, 32)

  // 2) Q/K/V projections + hnorm + pack (stream-ordered, yraw reused)
  k_gemm<<<ggrid, 256, 0, stream>>>(xh, Wqh, yraw);
  k_hnorm_pack<<<ROWS, 256, 0, stream>>>(yraw, bq, qh, nullptr, 0);
  k_gemm<<<ggrid, 256, 0, stream>>>(xh, Wkh, yraw);
  k_hnorm_pack<<<ROWS, 256, 0, stream>>>(yraw, bk, khb, nullptr, 0);
  k_gemm<<<ggrid, 256, 0, stream>>>(xh, Wvh, yraw);
  k_hnorm_pack<<<ROWS, 256, 0, stream>>>(yraw, bv, vtb, nullptr, 1);

  // 3) causal flash attention
  k_attn<<<dim3(SEQ / 64, BATCH * HEADS), 128, 0, stream>>>(qh, khb, vtb, ahb);

  // 4) output projection + hnorm -> fp32 d_out
  k_gemm<<<ggrid, 256, 0, stream>>>(ahb, Woh, yraw);
  k_hnorm_pack<<<ROWS, 256, 0, stream>>>(yraw, bo, nullptr, (float*)d_out, 2);
}
